// MultiHeadSelfAttention_80461917323589
// MI455X (gfx1250) — matrix operand order repaired and checked by reading, not compiled, not run on hardware
//
#include <hip/hip_runtime.h>
#include <hip/hip_bf16.h>
#include <stdint.h>

// ---------------------------------------------------------------------------
// MHA for gfx1250 (MI455X): fp32 in/out, bf16 WMMA compute (f32 accumulate).
//   D_MODEL=1024, H=16, Dh=64, B=2, S=2048.
// Pipeline: cvt f32->bf16 -> 3x proj GEMM (WMMA, async-LDS double buffered)
//           -> flash attention (WMMA, async-LDS K/V streaming)
//           -> output proj GEMM (WMMA, writes fp32 d_out).
// Workspace usage: 48 MB.
// ---------------------------------------------------------------------------

typedef __attribute__((ext_vector_type(16))) __bf16 v16bf;
typedef __attribute__((ext_vector_type(8)))  float  v8f;

#define D_MODEL  1024
#define NUM_HEADS 16
#define HEAD_DIM  64
#define SEQ     2048
#define BATCH      2
#define M_TOTAL (BATCH * SEQ)   /* 4096 */

// ---------------- gfx1250 async global->LDS copy (ASYNCcnt tracked) --------

static __device__ __forceinline__ void async_b128(const void* lds_dst,
                                                  uint32_t voff_bytes,
                                                  uint64_t sbase) {
  // GVS mode: mem_addr = SADDR + VADDR(int32); VDST = per-lane LDS byte addr.
  const uint32_t l = (uint32_t)(uintptr_t)lds_dst;
  asm volatile("global_load_async_to_lds_b128 %0, %1, %2"
               :: "v"(l), "v"(voff_bytes), "s"(sbase)
               : "memory");
}
#define WAIT_ASYNC_0()  asm volatile("s_wait_asynccnt 0x0" ::: "memory")
#define WAIT_ASYNC_16() asm volatile("s_wait_asynccnt 0x10" ::: "memory")

// ---------------- WMMA fragment helpers (wave32, A/B k = 16*(lane/16) + elem) -------

static __device__ __forceinline__ v8f wmma_bf16(v16bf a, v16bf b, v8f c) {
  return __builtin_amdgcn_wmma_f32_16x16x32_bf16(
      /*neg_a=*/false, a, /*neg_b=*/false, b,
      /*c_mod=*/(short)0, c, /*reuse_a=*/false, /*reuse_b=*/false);
}

// A-matrix 16x32 bf16 from row-major tile [16][ld].
static __device__ __forceinline__ v16bf load_a_frag(const __bf16* t, int ld,
                                                    int lane) {
  const int m  = lane & 15;
  const int hi = (lane >> 4) & 1;
  v16bf a;
#pragma unroll
  for (int v = 0; v < 8; ++v) {
    const int kb = 2 * v + 16 * hi;
    a[2 * v]     = t[m * ld + kb];
    a[2 * v + 1] = t[m * ld + kb + 1];
  }
  return a;
}

// B-matrix 32x16 bf16 from row-major B[k][n] (stride ld).
static __device__ __forceinline__ v16bf load_b_frag_rowmajor(const __bf16* B,
                                                             int ld, int lane) {
  const int n     = lane & 15;
  const int khalf = (lane >> 4) * 16;
  v16bf b;
#pragma unroll
  for (int v = 0; v < 8; ++v) {
    b[2 * v]     = B[(khalf + 2 * v) * ld + n];
    b[2 * v + 1] = B[(khalf + 2 * v + 1) * ld + n];
  }
  return b;
}

// B-matrix 32x16 bf16 from transposed storage Bt[n][k] (stride ld).
static __device__ __forceinline__ v16bf load_b_frag_nmajor(const __bf16* Bt,
                                                           int ld, int lane) {
  const int n     = lane & 15;
  const int khalf = (lane >> 4) * 16;
  v16bf b;
#pragma unroll
  for (int v = 0; v < 8; ++v) {
    b[2 * v]     = Bt[n * ld + khalf + 2 * v];
    b[2 * v + 1] = Bt[n * ld + khalf + 2 * v + 1];
  }
  return b;
}

// ---------------- fp32 -> bf16 convert (float4 -> 4x bf16) -----------------

__global__ void cvt_f32_bf16(const float4* __restrict__ in,
                             uint2* __restrict__ out, int n4) {
  const int i = blockIdx.x * blockDim.x + threadIdx.x;
  if (i < n4) {
    const float4 f = in[i];
    union { __bf16 h[4]; uint2 u; } cv;
    cv.h[0] = (__bf16)f.x; cv.h[1] = (__bf16)f.y;
    cv.h[2] = (__bf16)f.z; cv.h[3] = (__bf16)f.w;
    out[i] = cv.u;
  }
}

// ---------------- tiled GEMM: C[M,N] = A[M,K] @ W[N,K]^T --------------------
// Block tile 128x128x32, 128 threads = 4 waves (2x2), wave tile 64x64
// (16 WMMAs / K-step). Double-buffered LDS fed by async global->LDS b128.
// mode 0: scatter bf16 result to QKV layout [B*H, S, 64].
// mode 1: write fp32 row-major [M, N] (final output).

__global__ __launch_bounds__(128)
void gemm_bf16(const __bf16* __restrict__ A, const __bf16* __restrict__ W,
               __bf16* __restrict__ out_qkv, float* __restrict__ out_f32,
               int mode) {
  __shared__ __bf16 As[2][128 * 32];   // 8 KB per stage
  __shared__ __bf16 Ws[2][128 * 32];   // 8 KB per stage (n-major: Bt[n][k])

  const int tid  = threadIdx.x;
  const int lane = tid & 31;
  const int wave = tid >> 5;
  const int wm   = (wave & 1) * 64;   // 2 waves along M
  const int wn   = (wave >> 1) * 64;  // 2 waves along N
  const int m0   = blockIdx.x * 128;
  const int n0   = blockIdx.y * 128;

  const uint64_t baseA = (uint64_t)(uintptr_t)(A + (size_t)m0 * D_MODEL);
  const uint64_t baseW = (uint64_t)(uintptr_t)(W + (size_t)n0 * D_MODEL);

  // 128x32 bf16 tile = 512 b128 pieces; each of 128 threads issues 4 per mat.
  // piece idx = tid + i*128 -> row = idx>>2, quarter q = idx&3
  uint32_t offs[4];
  int      lidx[4];
#pragma unroll
  for (int i = 0; i < 4; ++i) {
    const int idx = tid + i * 128;
    lidx[i] = idx;
    offs[i] = (uint32_t)((idx >> 2) * (D_MODEL * 2) + (idx & 3) * 16);
  }

  auto issue = [&](int stage, int k0) {
    const uint32_t kb = (uint32_t)(k0 * 2);
#pragma unroll
    for (int i = 0; i < 4; ++i) {
      async_b128(&As[stage][lidx[i] * 8], offs[i] + kb, baseA);
      async_b128(&Ws[stage][lidx[i] * 8], offs[i] + kb, baseW);
    }
  };

  v8f acc[4][4];
  const v8f zero = {};
#pragma unroll
  for (int i = 0; i < 4; ++i)
#pragma unroll
    for (int j = 0; j < 4; ++j) acc[i][j] = zero;

  issue(0, 0);
  WAIT_ASYNC_0();
  __syncthreads();

  const int NSTEP = D_MODEL / 32;  // 32
#pragma unroll 2
  for (int step = 0; step < NSTEP; ++step) {
    const int cur = step & 1;
    if (step + 1 < NSTEP) issue(cur ^ 1, (step + 1) * 32);

    const __bf16* Asb = As[cur];
    const __bf16* Bsb = Ws[cur];
    v16bf af[4], bfr[4];
#pragma unroll
    for (int t = 0; t < 4; ++t)
      af[t] = load_a_frag(Asb + (wm + t * 16) * 32, 32, lane);
#pragma unroll
    for (int t = 0; t < 4; ++t)
      bfr[t] = load_b_frag_nmajor(Bsb + (wn + t * 16) * 32, 32, lane);

#pragma unroll
    for (int mt = 0; mt < 4; ++mt)
#pragma unroll
      for (int nt = 0; nt < 4; ++nt)
        acc[mt][nt] = wmma_bf16(af[mt], bfr[nt], acc[mt][nt]);

    if (step + 1 < NSTEP) WAIT_ASYNC_0();
    __syncthreads();
  }

  // C layout (f32 16x16): lane n=lane&15, VGPR r -> M=r (lo half) / r+8 (hi).
  const int cn = lane & 15;
  const int hi = (lane >> 4) & 1;
#pragma unroll
  for (int mt = 0; mt < 4; ++mt)
#pragma unroll
    for (int nt = 0; nt < 4; ++nt)
#pragma unroll
      for (int r = 0; r < 8; ++r) {
        const int gm = m0 + wm + mt * 16 + r + 8 * hi;
        const int gn = n0 + wn + nt * 16 + cn;
        const float val = acc[mt][nt][r];
        if (mode == 0) {
          const int b_ = gm >> 11;
          const int s_ = gm & (SEQ - 1);
          const int h_ = gn >> 6;
          const int d_ = gn & (HEAD_DIM - 1);
          out_qkv[(((size_t)(b_ * NUM_HEADS + h_)) * SEQ + s_) * HEAD_DIM + d_] =
              (__bf16)val;
        } else {
          out_f32[(size_t)gm * D_MODEL + gn] = val;
        }
      }
}

// ---------------- causal flash attention ------------------------------------
// Grid (B*H, S/16); one wave = 16 queries x Dh=64, streaming 32-key blocks
// through double-buffered async-LDS. 8 WMMAs per key block.

__global__ __launch_bounds__(32)
void attn_causal(const __bf16* __restrict__ Q, const __bf16* __restrict__ K,
                 const __bf16* __restrict__ V, __bf16* __restrict__ O) {
  __shared__ __bf16 Qs[16 * 64];       // 2 KB
  __shared__ __bf16 Ks[2][32 * 64];    // 4 KB per stage
  __shared__ __bf16 Vs[2][32 * 64];    // 4 KB per stage
  __shared__ __bf16 Ps[16 * 32];       // probs tile

  const int lane = threadIdx.x;
  const int bh   = blockIdx.x;
  const int q0   = blockIdx.y * 16;
  const int n    = lane & 15;
  const int hi   = (lane >> 4) & 1;

  const uint64_t baseQ = (uint64_t)(uintptr_t)(Q + (size_t)(bh * SEQ + q0) * HEAD_DIM);
  const uint64_t baseK = (uint64_t)(uintptr_t)(K + (size_t)bh * SEQ * HEAD_DIM);
  const uint64_t baseV = (uint64_t)(uintptr_t)(V + (size_t)bh * SEQ * HEAD_DIM);

  // Q tile: 16x64 bf16 = 128 b128 pieces -> 4 per lane
#pragma unroll
  for (int i = 0; i < 4; ++i)
    async_b128(&Qs[(lane + i * 32) * 8], (uint32_t)((lane + i * 32) * 16), baseQ);

  // K/V block: 32x64 bf16 = 256 b128 pieces -> 8 per lane each (16 asyncs)
  auto issue_kv = [&](int stage, int k0) {
    const uint32_t kb = (uint32_t)(k0 * HEAD_DIM * 2);
#pragma unroll
    for (int i = 0; i < 8; ++i) {
      const uint32_t o = (uint32_t)((lane + i * 32) * 16);
      async_b128(&Ks[stage][(lane + i * 32) * 8], kb + o, baseK);
      async_b128(&Vs[stage][(lane + i * 32) * 8], kb + o, baseV);
    }
  };

  issue_kv(0, 0);
  WAIT_ASYNC_16();  // Q (oldest 4) complete; K/V tile 0 still in flight
  const v16bf aq0 = load_a_frag(Qs, HEAD_DIM, lane);
  const v16bf aq1 = load_a_frag(Qs + 32, HEAD_DIM, lane);

  const v8f zero = {};
  v8f acc[4];
#pragma unroll
  for (int t = 0; t < 4; ++t) acc[t] = zero;
  float mrow[8], lrow[8];
#pragma unroll
  for (int r = 0; r < 8; ++r) { mrow[r] = -1e30f; lrow[r] = 0.0f; }

  const int nkb = (q0 >> 5) + 1;
  for (int kbi = 0; kbi < nkb; ++kbi) {
    const int k0  = kbi * 32;
    const int cur = kbi & 1;
    if (kbi + 1 < nkb) {
      issue_kv(cur ^ 1, k0 + 32);
      WAIT_ASYNC_16();  // retire current tile's 16 (in-order completion)
    } else {
      WAIT_ASYNC_0();
    }
    const __bf16* Kc = Ks[cur];
    const __bf16* Vc = Vs[cur];

    // scores = Q (16x64) @ K^T (64x32): Bt[n=key][k=dim] == K rows
    v8f s0 = wmma_bf16(aq0, load_b_frag_nmajor(Kc, HEAD_DIM, lane), zero);
    s0     = wmma_bf16(aq1, load_b_frag_nmajor(Kc + 32, HEAD_DIM, lane), s0);
    v8f s1 = wmma_bf16(aq0, load_b_frag_nmajor(Kc + 16 * HEAD_DIM, HEAD_DIM, lane), zero);
    s1     = wmma_bf16(aq1, load_b_frag_nmajor(Kc + 16 * HEAD_DIM + 32, HEAD_DIM, lane), s1);

    // online softmax; row M of the C tile lives across 16 lanes in VGPR r
#pragma unroll
    for (int r = 0; r < 8; ++r) {
      const int qrow = q0 + r + 8 * hi;
      float v0 = ((k0 + n)      <= qrow) ? s0[r] * 0.125f : -1e30f;
      float v1 = ((k0 + 16 + n) <= qrow) ? s1[r] * 0.125f : -1e30f;
      float mr = fmaxf(v0, v1);
#pragma unroll
      for (int off = 8; off >= 1; off >>= 1)
        mr = fmaxf(mr, __shfl_xor(mr, off, 32));
      const float mnew  = fmaxf(mrow[r], mr);
      const float alpha = __expf(mrow[r] - mnew);
      const float p0 = __expf(v0 - mnew);
      const float p1 = __expf(v1 - mnew);
      float rs = p0 + p1;
#pragma unroll
      for (int off = 8; off >= 1; off >>= 1) rs += __shfl_xor(rs, off, 32);
      lrow[r] = lrow[r] * alpha + rs;
      mrow[r] = mnew;
#pragma unroll
      for (int t = 0; t < 4; ++t) acc[t][r] *= alpha;
      Ps[(r + 8 * hi) * 32 + n]      = (__bf16)p0;
      Ps[(r + 8 * hi) * 32 + 16 + n] = (__bf16)p1;
    }
    __syncthreads();  // LDS ordering for Ps within the wave

    const v16bf pa = load_a_frag(Ps, 32, lane);
#pragma unroll
    for (int t = 0; t < 4; ++t)
      acc[t] = wmma_bf16(pa, load_b_frag_rowmajor(Vc + t * 16, HEAD_DIM, lane),
                         acc[t]);
  }

  const int b_ = bh >> 4;
  const int h_ = bh & 15;
#pragma unroll
  for (int t = 0; t < 4; ++t)
#pragma unroll
    for (int r = 0; r < 8; ++r) {
      const int row = q0 + r + 8 * hi;
      const int col = h_ * HEAD_DIM + t * 16 + n;
      O[((size_t)(b_ * SEQ) + row) * D_MODEL + col] =
          (__bf16)(acc[t][r] / lrow[r]);
    }
}

// ---------------- host launcher ---------------------------------------------

extern "C" void kernel_launch(void* const* d_in, const int* in_sizes, int n_in,
                              void* d_out, int out_size, void* d_ws,
                              size_t ws_size, hipStream_t stream) {
  const float* wq = (const float*)d_in[0];
  const float* wk = (const float*)d_in[1];
  const float* wv = (const float*)d_in[2];
  const float* wo = (const float*)d_in[3];
  const float* x  = (const float*)d_in[4];

  char* ws = (char*)d_ws;   // needs 48 MB
  __bf16* xbf = (__bf16*)(ws);                         // 8 MB  [4096,1024]
  __bf16* wqb = (__bf16*)(ws + ((size_t)8  << 20));
  __bf16* wkb = (__bf16*)(ws + ((size_t)10 << 20));
  __bf16* wvb = (__bf16*)(ws + ((size_t)12 << 20));
  __bf16* wob = (__bf16*)(ws + ((size_t)14 << 20));
  __bf16* qb  = (__bf16*)(ws + ((size_t)16 << 20));    // 8 MB [32,2048,64]
  __bf16* kb  = (__bf16*)(ws + ((size_t)24 << 20));
  __bf16* vb  = (__bf16*)(ws + ((size_t)32 << 20));
  __bf16* mg  = (__bf16*)(ws + ((size_t)40 << 20));    // 8 MB [2,2048,1024]

  const int nX4 = (M_TOTAL * D_MODEL) / 4;
  const int nW4 = (D_MODEL * D_MODEL) / 4;
  cvt_f32_bf16<<<(nX4 + 255) / 256, 256, 0, stream>>>((const float4*)x,  (uint2*)xbf, nX4);
  cvt_f32_bf16<<<(nW4 + 255) / 256, 256, 0, stream>>>((const float4*)wq, (uint2*)wqb, nW4);
  cvt_f32_bf16<<<(nW4 + 255) / 256, 256, 0, stream>>>((const float4*)wk, (uint2*)wkb, nW4);
  cvt_f32_bf16<<<(nW4 + 255) / 256, 256, 0, stream>>>((const float4*)wv, (uint2*)wvb, nW4);
  cvt_f32_bf16<<<(nW4 + 255) / 256, 256, 0, stream>>>((const float4*)wo, (uint2*)wob, nW4);

  const dim3 gGemm(M_TOTAL / 128, D_MODEL / 128);      // 32 x 8
  gemm_bf16<<<gGemm, 128, 0, stream>>>(xbf, wqb, qb, nullptr, 0);
  gemm_bf16<<<gGemm, 128, 0, stream>>>(xbf, wkb, kb, nullptr, 0);
  gemm_bf16<<<gGemm, 128, 0, stream>>>(xbf, wvb, vb, nullptr, 0);

  const dim3 gAttn(BATCH * NUM_HEADS, SEQ / 16);       // 32 x 128
  attn_causal<<<gAttn, 32, 0, stream>>>(qb, kb, vb, mg);

  gemm_bf16<<<gGemm, 128, 0, stream>>>(mg, wob, nullptr, (float*)d_out, 1);
}
